// AttentionHead_27479200760152
// MI455X (gfx1250) — matrix operand order repaired and checked by reading, not compiled, not run on hardware
//
#include <hip/hip_runtime.h>
#include <math.h>

typedef __attribute__((ext_vector_type(16))) __bf16 v16bf;
typedef __attribute__((ext_vector_type(8)))  __bf16 bf8;
typedef __attribute__((ext_vector_type(8)))  float  v8f;
typedef __attribute__((ext_vector_type(4)))  float  f4;

#define DMODEL 512
#define DKV    64
#define NB     8
#define NSEQ   4096

// ---------------------------------------------------------------------------
// CDNA5 async copy: global -> LDS (16B per lane), tracked by ASYNCcnt.
// ---------------------------------------------------------------------------
__device__ __forceinline__ void async_copy16(void* lds, const void* gptr) {
    unsigned ldsa = (unsigned)(size_t)lds;                 // low 32 bits = LDS byte addr
    unsigned long long ga = (unsigned long long)(size_t)gptr;
    asm volatile("global_load_async_to_lds_b128 %0, %1, off"
                 :: "v"(ldsa), "v"(ga) : "memory");
}
__device__ __forceinline__ void wait_async_le(int n) {
    if (n == 0) asm volatile("s_wait_asynccnt 0x0" ::: "memory");
    else        asm volatile("s_wait_asynccnt 0x2" ::: "memory");
}

// ---------------------------------------------------------------------------
// Projection: Out(bf16) = X(f32, rows x 512) @ W(f32, 512x64) + b
// One wave computes a 16x64 tile via v_wmma_f32_16x16x32_bf16.
// transposeOut=0: Out[row][col] (row-major [b*n][64])
// transposeOut=1: Out[b][col][n] ([b][64][4096]) -- used for V so the
//                 attention kernel can async-copy B-operand-ready tiles.
// ---------------------------------------------------------------------------
__global__ __launch_bounds__(256) void proj_kernel(const float* __restrict__ X,
                                                   const float* __restrict__ W,
                                                   const float* __restrict__ bias,
                                                   unsigned short* __restrict__ OutUS,
                                                   int transposeOut)
{
    __shared__ alignas(16) __bf16 Wt[DKV * DMODEL];   // [64][512] bf16, 64KB
    __bf16* Out = (__bf16*)OutUS;

    const int tid = threadIdx.x;
    for (int i = tid; i < DMODEL * DKV; i += 256) {
        int r = i >> 6, c = i & 63;
        Wt[c * DMODEL + r] = (__bf16)W[i];
    }
    __syncthreads();

    const int wave  = tid >> 5;
    const int lane  = tid & 31;
    const int half  = lane >> 4;
    const int col16 = lane & 15;
    const int g0    = half * 8;
    const int row0  = (blockIdx.x * 8 + wave) * 16;
    const int arow  = row0 + col16;

    v8f acc[4] = {};
    for (int k0 = 0; k0 < DMODEL; k0 += 32) {
        const float* ap = X + (size_t)arow * DMODEL + k0;
        f4 a0 = *(const f4*)(ap + g0);
        f4 a1 = *(const f4*)(ap + g0 + 4);
        f4 a2 = *(const f4*)(ap + 16 + g0);
        f4 a3 = *(const f4*)(ap + 16 + g0 + 4);
        v16bf A;
#pragma unroll
        for (int i = 0; i < 4; ++i) {
            A[i]      = (__bf16)a0[i];
            A[4 + i]  = (__bf16)a1[i];
            A[8 + i]  = (__bf16)a2[i];
            A[12 + i] = (__bf16)a3[i];
        }
#pragma unroll
        for (int nt = 0; nt < 4; ++nt) {
            const __bf16* bp = &Wt[(nt * 16 + col16) * DMODEL + k0 + half * 8];
            bf8 b0 = *(const bf8*)bp;
            bf8 b1 = *(const bf8*)(bp + 16);
            v16bf B;
#pragma unroll
            for (int i = 0; i < 8; ++i) { B[i] = b0[i]; B[8 + i] = b1[i]; }
            acc[nt] = __builtin_amdgcn_wmma_f32_16x16x32_bf16(
                false, A, false, B, (short)0, acc[nt], false, false);
        }
    }

#pragma unroll
    for (int nt = 0; nt < 4; ++nt) {
        int col  = nt * 16 + col16;
        float bv = bias[col];
#pragma unroll
        for (int r = 0; r < 8; ++r) {
            int row = row0 + half * 8 + r;
            __bf16 val = (__bf16)(acc[nt][r] + bv);
            if (!transposeOut) {
                Out[(size_t)row * DKV + col] = val;
            } else {
                size_t b = row >> 12, n = row & (NSEQ - 1);
                Out[(b * DKV + col) * NSEQ + n] = val;
            }
        }
    }
}

// ---------------------------------------------------------------------------
// Flash attention. Block = 8 waves, each owning a 16-row Q tile; waves share
// double-buffered 32-key K/V LDS tiles filled by global_load_async_to_lds.
// ---------------------------------------------------------------------------
__global__ __launch_bounds__(256) void attn_kernel(const unsigned short* __restrict__ QpUS,
                                                   const unsigned short* __restrict__ KpUS,
                                                   const unsigned short* __restrict__ VtUS,
                                                   const int* __restrict__ maskPtr,
                                                   float* __restrict__ Out)
{
    __shared__ alignas(16) __bf16 ktile[2][32 * 64];   // [key][dk]   4KB x2
    __shared__ alignas(16) __bf16 vtile[2][64 * 32];   // [dv][key]   4KB x2
    __shared__ alignas(16) __bf16 pS[8][16 * 32];      // per-wave P  8KB

    const __bf16* Qp = (const __bf16*)QpUS;
    const __bf16* Kp = (const __bf16*)KpUS;
    const __bf16* Vt = (const __bf16*)VtUS;

    const int tid   = threadIdx.x;
    const int wave  = tid >> 5;
    const int lane  = tid & 31;
    const int half  = lane >> 4;
    const int col16 = lane & 15;
    const int g0    = half * 8;
    const int batch = blockIdx.y;
    const int qrow0 = (blockIdx.x * 8 + wave) * 16;
    const float maskv = (float)maskPtr[0];
    const size_t base = (size_t)batch * NSEQ * DKV;    // same flat base for Kp and Vt

    // Issue one 16B async chunk per thread per matrix (1 wave-instr each).
    auto issue_tiles = [&](int buf, int j0) {
        async_copy16(&ktile[buf][tid * 8],
                     Kp + base + (size_t)j0 * DKV + tid * 8);
        async_copy16(&vtile[buf][tid * 8],
                     Vt + base + (size_t)(tid >> 2) * NSEQ + j0 + (tid & 3) * 8);
    };

    // Q as two 16x32 A tiles, resident for all 128 key steps
    v16bf QA[2];
    {
        const __bf16* qp = Qp + base + (size_t)(qrow0 + col16) * DKV;
#pragma unroll
        for (int t = 0; t < 2; ++t) {
            bf8 lo = *(const bf8*)(qp + t * 32 + g0);
            bf8 hi = *(const bf8*)(qp + t * 32 + 16 + g0);
#pragma unroll
            for (int i = 0; i < 8; ++i) { QA[t][i] = lo[i]; QA[t][8 + i] = hi[i]; }
        }
    }

    v8f o[4] = {};
    float m[8], l[8];
#pragma unroll
    for (int r = 0; r < 8; ++r) { m[r] = -INFINITY; l[r] = 0.f; }

    issue_tiles(0, 0);
    issue_tiles(1, 32);

    for (int j0 = 0; j0 < NSEQ; j0 += 32) {
        const int buf = (j0 >> 5) & 1;
        wait_async_le(2);          // oldest tile-set (this buf) landed in LDS
        __syncthreads();           // all waves' async chunks visible

        // ---- S = Q K^T : two 16x16 score tiles, K-dim 64 = 2 WMMA each
        v8f c[2];
#pragma unroll
        for (int st = 0; st < 2; ++st) {
            v8f s = {};
            const __bf16* kpr = &ktile[buf][(st * 16 + col16) * DKV];
#pragma unroll
            for (int t = 0; t < 2; ++t) {
                const __bf16* bp = kpr + t * 32 + half * 8;
                bf8 b0 = *(const bf8*)bp;
                bf8 b1 = *(const bf8*)(bp + 16);
                v16bf B;
#pragma unroll
                for (int i = 0; i < 8; ++i) { B[i] = b0[i]; B[8 + i] = b1[i]; }
                s = __builtin_amdgcn_wmma_f32_16x16x32_bf16(
                        false, QA[t], false, B, (short)0, s, false, false);
            }
            c[st] = s;
        }

        // ---- online softmax (row stats live across each 16-lane half)
        float p0[8], p1[8], alpha[8];
#pragma unroll
        for (int r = 0; r < 8; ++r) {
            float s0 = c[0][r] * 0.125f;           // /sqrt(64)
            float s1 = c[1][r] * 0.125f;
            if (s0 == maskv) s0 = -INFINITY;       // attn == mask -> -inf
            if (s1 == maskv) s1 = -INFINITY;
            float mx = fmaxf(s0, s1);
            mx = fmaxf(mx, __shfl_xor(mx, 1));
            mx = fmaxf(mx, __shfl_xor(mx, 2));
            mx = fmaxf(mx, __shfl_xor(mx, 4));
            mx = fmaxf(mx, __shfl_xor(mx, 8));
            float mnew = fmaxf(m[r], mx);
            float a  = __expf(m[r] - mnew);
            p0[r] = __expf(s0 - mnew);
            p1[r] = __expf(s1 - mnew);
            float rs = p0[r] + p1[r];
            rs += __shfl_xor(rs, 1);
            rs += __shfl_xor(rs, 2);
            rs += __shfl_xor(rs, 4);
            rs += __shfl_xor(rs, 8);
            l[r] = l[r] * a + rs;
            m[r] = mnew;
            alpha[r] = a;
        }
#pragma unroll
        for (int nt = 0; nt < 4; ++nt)
#pragma unroll
            for (int r = 0; r < 8; ++r) o[nt][r] *= alpha[r];

        // ---- P: C layout -> A layout via per-wave LDS staging
        __bf16* pw = pS[wave];
#pragma unroll
        for (int r = 0; r < 8; ++r) {
            int row = half * 8 + r;
            pw[row * 32 + col16]      = (__bf16)p0[r];
            pw[row * 32 + 16 + col16] = (__bf16)p1[r];
        }
        __syncthreads();

        v16bf PA;
        {
            const __bf16* pr = pw + col16 * 32;
            bf8 lo = *(const bf8*)(pr + g0);
            bf8 hi = *(const bf8*)(pr + 16 + g0);
#pragma unroll
            for (int i = 0; i < 8; ++i) { PA[i] = lo[i]; PA[8 + i] = hi[i]; }
        }

        // ---- O += P @ V : B-operand rows contiguous in [dv][key] tile
#pragma unroll
        for (int nt = 0; nt < 4; ++nt) {
            const __bf16* vp = &vtile[buf][(nt * 16 + col16) * 32 + half * 8];
            bf8 b0 = *(const bf8*)vp;
            bf8 b1 = *(const bf8*)(vp + 16);
            v16bf B;
#pragma unroll
            for (int i = 0; i < 8; ++i) { B[i] = b0[i]; B[8 + i] = b1[i]; }
            o[nt] = __builtin_amdgcn_wmma_f32_16x16x32_bf16(
                        false, PA, false, B, (short)0, o[nt], false, false);
        }

        __syncthreads();           // all waves done reading this buffer
        if (j0 + 64 < NSEQ) issue_tiles(buf, j0 + 64);
    }

    // ---- epilogue: divide by row sums, write fp32
#pragma unroll
    for (int nt = 0; nt < 4; ++nt) {
        int col = nt * 16 + col16;
#pragma unroll
        for (int r = 0; r < 8; ++r) {
            int row = qrow0 + half * 8 + r;
            float denom = l[r];
            Out[base + (size_t)row * DKV + col] = (denom > 0.f) ? o[nt][r] / denom : 0.f;
        }
    }
}

// ---------------------------------------------------------------------------
extern "C" void kernel_launch(void* const* d_in, const int* in_sizes, int n_in,
                              void* d_out, int out_size, void* d_ws, size_t ws_size,
                              hipStream_t stream)
{
    const float* q  = (const float*)d_in[0];
    const float* k  = (const float*)d_in[1];
    const float* v  = (const float*)d_in[2];
    const float* Wq = (const float*)d_in[3];
    const float* bq = (const float*)d_in[4];
    const float* Wk = (const float*)d_in[5];
    const float* bk = (const float*)d_in[6];
    const float* Wv = (const float*)d_in[7];
    const float* bv = (const float*)d_in[8];
    const int*   mk = (const int*)d_in[9];

    const size_t elems = (size_t)NB * NSEQ * DKV;
    unsigned short* qp = (unsigned short*)d_ws;
    unsigned short* kp = qp + elems;
    unsigned short* vt = kp + elems;          // [b][64][4096]

    dim3 pb(256), pg(256);
    proj_kernel<<<pg, pb, 0, stream>>>(q, Wq, bq, qp, 0);
    proj_kernel<<<pg, pb, 0, stream>>>(k, Wk, bk, kp, 0);
    proj_kernel<<<pg, pb, 0, stream>>>(v, Wv, bv, vt, 1);

    attn_kernel<<<dim3(32, NB), 256, 0, stream>>>(qp, kp, vt, mk, (float*)d_out);
}
